// RBF_Softmax_Layer_81088982548814
// MI455X (gfx1250) — matrix operand order repaired and checked by reading, not compiled, not run on hardware
//
#include <hip/hip_runtime.h>
#include <hip/hip_bf16.h>

#define B_ROWS 4096
#define D_DIM  512
#define C_DIM  32000
#define NSUB   (C_DIM / 64)   // 500 per-row partial-sum slots (one per 64-col wave tile)

typedef __attribute__((ext_vector_type(16))) __bf16       v16bf;
typedef __attribute__((ext_vector_type(8)))  float        v8f;
typedef __attribute__((ext_vector_type(4)))  float        v4f;
typedef __attribute__((ext_vector_type(4)))  unsigned int v4u;

// ---------------- Kernel 1: L2-normalize rows of x -> e (bf16) ----------------
__global__ void row_normalize_kernel(const float* __restrict__ x, __bf16* __restrict__ e) {
    __shared__ float red[256];
    const int row = blockIdx.x;
    const int t   = threadIdx.x;
    const float* xr = x + (size_t)row * D_DIM;
    float v0 = xr[t], v1 = xr[t + 256];
    red[t] = v0 * v0 + v1 * v1;
    __syncthreads();
    for (int off = 128; off > 0; off >>= 1) {
        if (t < off) red[t] += red[t + off];
        __syncthreads();
    }
    float rinv = rsqrtf(fmaxf(red[0], 1e-12f));
    __bf16* er = e + (size_t)row * D_DIM;
    er[t]       = (__bf16)(v0 * rinv);
    er[t + 256] = (__bf16)(v1 * rinv);
}

// ---------------- Kernel 2a: column inverse norms of class kernel ----------------
__global__ void col_norm_kernel(const float* __restrict__ k, float* __restrict__ colinv) {
    const int c = blockIdx.x * 256 + threadIdx.x;
    float s = 0.0f;
    for (int d = 0; d < D_DIM; ++d) {
        float v = k[(size_t)d * C_DIM + c];
        s += v * v;
    }
    colinv[c] = rsqrtf(fmaxf(s, 1e-12f));
}

// ---------------- Kernel 2b: transpose + scale -> wt[C][D] bf16 ----------------
__global__ void transpose_scale_kernel(const float* __restrict__ k,
                                       const float* __restrict__ colinv,
                                       __bf16* __restrict__ wt) {
    __shared__ float tile[32][33];
    const int tx = threadIdx.x, ty = threadIdx.y;
    const int c0 = blockIdx.x * 32, d0 = blockIdx.y * 32;
#pragma unroll
    for (int j = 0; j < 4; ++j)
        tile[ty + 8 * j][tx] = k[(size_t)(d0 + ty + 8 * j) * C_DIM + (c0 + tx)];
    __syncthreads();
#pragma unroll
    for (int j = 0; j < 4; ++j) {
        int c = c0 + ty + 8 * j;
        wt[(size_t)c * D_DIM + (d0 + tx)] = (__bf16)(tile[tx][ty + 8 * j] * colinv[c]);
    }
}

// ---------------- Kernel 3: WMMA GEMM + fused RBF/exp epilogue ----------------
// Block: 256 threads = 8 waves in a 2(M) x 4(N) grid. Block tile 64 x 256.
// Wave tile: 32 rows x 64 cols = 2 A-frags x 4 B-frags = 8 WMMA accumulators.
// All loads use 3 per-lane base pointers + compile-time immediate offsets;
// K-loop is double-buffered so next step's 12 loads overlap current 8 WMMAs.
__global__ void __launch_bounds__(256)
rbf_gemm_kernel(const __bf16* __restrict__ e,
                const __bf16* __restrict__ wt,
                float* __restrict__ out,
                float* __restrict__ partial) {
    const int lane = threadIdx.x & 31;
    const int wave = threadIdx.x >> 5;
    const int wm   = wave >> 2;        // 0..1
    const int wn   = wave & 3;         // 0..3
    const int half = lane >> 4;        // 0..1
    const int l16  = lane & 15;
    const int m0 = blockIdx.y * 64 + wm * 32;
    const int n0 = blockIdx.x * 256 + wn * 64;

    v8f acc[2][4];
#pragma unroll
    for (int mi = 0; mi < 2; ++mi)
#pragma unroll
        for (int t = 0; t < 4; ++t)
            acc[mi][t] = (v8f){};

    // Per-lane byte base pointers; every load below is base + immediate offset.
    const char* aB0 = (const char*)e  + ((size_t)(m0 + l16) * D_DIM + 8 * half) * 2;
    const char* aB1 = (const char*)e  + ((size_t)(m0 + 16 + l16) * D_DIM + 8 * half) * 2;
    const char* wB  = (const char*)wt + ((size_t)(n0 + l16) * D_DIM + 8 * half) * 2;

    v16bf a0[2], a1[2], b[2][4];

#define LOAD_A(buf, kk)                                                   \
    do {                                                                  \
        ((v4u*)&a0[buf])[0] = *(const v4u*)(aB0 + (kk) * 64);             \
        ((v4u*)&a0[buf])[1] = *(const v4u*)(aB0 + (kk) * 64 + 32);        \
        ((v4u*)&a1[buf])[0] = *(const v4u*)(aB1 + (kk) * 64);             \
        ((v4u*)&a1[buf])[1] = *(const v4u*)(aB1 + (kk) * 64 + 32);        \
    } while (0)

#define LOAD_B(buf, kk)                                                   \
    do {                                                                  \
        _Pragma("unroll")                                                 \
        for (int t = 0; t < 4; ++t) {                                     \
            ((v4u*)&b[buf][t])[0] =                                       \
                *(const v4u*)(wB + t * 16384 + (kk) * 64);                \
            ((v4u*)&b[buf][t])[1] =                                       \
                *(const v4u*)(wB + t * 16384 + (kk) * 64 + 32);           \
        }                                                                 \
    } while (0)

    LOAD_A(0, 0);
    LOAD_B(0, 0);
#pragma unroll
    for (int kk = 0; kk < 16; ++kk) {
        const int cur = kk & 1;
        const int nxt = cur ^ 1;
        if (kk < 15) {          // prefetch next K-step while current WMMAs run
            LOAD_A(nxt, kk + 1);
            LOAD_B(nxt, kk + 1);
        }
#pragma unroll
        for (int t = 0; t < 4; ++t) {
            acc[0][t] = __builtin_amdgcn_wmma_f32_16x16x32_bf16(
                false, a0[cur], false, b[cur][t], (short)0, acc[0][t], false, false);
            acc[1][t] = __builtin_amdgcn_wmma_f32_16x16x32_bf16(
                false, a1[cur], false, b[cur][t], (short)0, acc[1][t], false, false);
        }
    }
#undef LOAD_A
#undef LOAD_B

    // Epilogue: logits = 100*exp((dot-1)/10); p = exp(logit - 100)  (shift by max=100)
#pragma unroll
    for (int mi = 0; mi < 2; ++mi) {
        float rowpart[8];
#pragma unroll
        for (int r = 0; r < 8; ++r) rowpart[r] = 0.0f;
#pragma unroll
        for (int t = 0; t < 4; ++t) {
            const int col = n0 + t * 16 + l16;
#pragma unroll
            for (int r = 0; r < 8; ++r) {
                float dot   = acc[mi][t][r];
                float logit = 100.0f * __expf((dot - 1.0f) * 0.1f);
                float p     = __expf(logit - 100.0f);
                out[(size_t)(m0 + mi * 16 + r + 8 * half) * C_DIM + col] = p;
                rowpart[r] += p;
            }
        }
        // Row partial sums: butterfly across the 16 lanes of each half-wave (= N dim)
#pragma unroll
        for (int r = 0; r < 8; ++r) {
            float s = rowpart[r];
            s += __shfl_xor(s, 1, 32);
            s += __shfl_xor(s, 2, 32);
            s += __shfl_xor(s, 4, 32);
            s += __shfl_xor(s, 8, 32);
            if (l16 == 0) {
                int row = m0 + mi * 16 + r + 8 * half;
                partial[(size_t)row * NSUB + (blockIdx.x * 4 + wn)] = s;
            }
        }
    }
}

// ---------------- Kernel 4: reduce per-row partials -> 1/sum ----------------
__global__ void rowsum_kernel(const float* __restrict__ partial, float* __restrict__ rowinv) {
    __shared__ float red[256];
    const int row = blockIdx.x;
    const float* pr = partial + (size_t)row * NSUB;
    float s = 0.0f;
    for (int i = threadIdx.x; i < NSUB; i += 256) s += pr[i];
    red[threadIdx.x] = s;
    __syncthreads();
    for (int off = 128; off > 0; off >>= 1) {
        if (threadIdx.x < off) red[threadIdx.x] += red[threadIdx.x + off];
        __syncthreads();
    }
    if (threadIdx.x == 0) rowinv[row] = 1.0f / red[0];
}

// ---------------- Kernel 5: scale out by 1/rowsum (vectorized) ----------------
__global__ void normalize_kernel(float* __restrict__ out, const float* __restrict__ rowinv) {
    size_t idx = ((size_t)blockIdx.x * 256 + threadIdx.x) * 4;
    int row = (int)(idx / C_DIM);           // C_DIM % 4 == 0, groups never cross rows
    float s = rowinv[row];
    v4f v = *(v4f*)(out + idx);
    v *= s;
    *(v4f*)(out + idx) = v;
}

extern "C" void kernel_launch(void* const* d_in, const int* in_sizes, int n_in,
                              void* d_out, int out_size, void* d_ws, size_t ws_size,
                              hipStream_t stream) {
    const float* x    = (const float*)d_in[0];   // [4096, 512]
    const float* kern = (const float*)d_in[1];   // [512, 32000]
    float* out = (float*)d_out;                  // [4096, 32000]

    char* ws = (char*)d_ws;
    size_t off = 0;
    __bf16* e       = (__bf16*)(ws + off); off += (size_t)B_ROWS * D_DIM * 2;      // 4 MB
    __bf16* wt      = (__bf16*)(ws + off); off += (size_t)C_DIM * D_DIM * 2;       // 32.8 MB
    float*  colinv  = (float*)(ws + off);  off += (size_t)C_DIM * 4;               // 128 KB
    float*  partial = (float*)(ws + off);  off += (size_t)B_ROWS * NSUB * 4;       // 8.2 MB
    float*  rowinv  = (float*)(ws + off);                                          // 16 KB

    row_normalize_kernel<<<B_ROWS, 256, 0, stream>>>(x, e);
    col_norm_kernel<<<C_DIM / 256, 256, 0, stream>>>(kern, colinv);
    transpose_scale_kernel<<<dim3(C_DIM / 32, D_DIM / 32), dim3(32, 8), 0, stream>>>(kern, colinv, wt);
    rbf_gemm_kernel<<<dim3(C_DIM / 256, B_ROWS / 64), 256, 0, stream>>>(e, wt, out, partial);
    rowsum_kernel<<<B_ROWS, 256, 0, stream>>>(partial, rowinv);
    normalize_kernel<<<(unsigned)((size_t)B_ROWS * C_DIM / 4 / 256), 256, 0, stream>>>(out, rowinv);
}
